// MCDropoutLSTM_9723805958617
// MI455X (gfx1250) — matrix operand order repaired and checked by reading, not compiled, not run on hardware
//
#include <hip/hip_runtime.h>
#include <hip/hip_bf16.h>

// ---------------------------------------------------------------------------
// MC-Dropout 2-layer LSTM encoder + 12-step decoder, fused for MI455X gfx1250.
// bf16 WMMA (v_wmma_f32_16x16x32_bf16) with f32 accumulation; wave32 layouts
// per CDNA5 ISA 7.12.2. One workgroup (512 threads / 16 waves) owns 64 batch
// rows end-to-end; each wave owns a 16-row x 2-gate-column-tile slice so cell
// state + accumulators + A-fragments stay well under 256 VGPRs (no spills).
// ---------------------------------------------------------------------------

typedef __attribute__((ext_vector_type(16))) __bf16       v16bf;
typedef __attribute__((ext_vector_type(8)))  float        v8f;
typedef __attribute__((ext_vector_type(4)))  unsigned int u32x4;

#define NB    32768   // batch
#define NT    32      // encoder time steps
#define ND    2       // obs feature dim
#define NH    128     // hidden
#define NG    512     // 4*NH gates
#define NPL   12      // decoder steps
#define NO    2       // output dim
#define BT    64      // batch rows per workgroup
#define NTHR  512     // 16 waves (wave32)
#define NJB   2       // gate-column tiles per wave

constexpr float kP    = 0.3f;
constexpr float kPinv = 1.0f / 0.7f;

struct U32x4Pair { u32x4 lo; u32x4 hi; };

__device__ __forceinline__ v16bf pack_bf(u32x4 a, u32x4 b) {
  U32x4Pair p{a, b};
  return __builtin_bit_cast(v16bf, p);
}

__device__ __forceinline__ float sig_(float x) { return 1.0f / (1.0f + __expf(-x)); }

// A-matrix fragment (16x32 bf16): lane&15 = row M; lane half selects K groups
// {0..7,16..23} vs {8..15,24..31}  (ISA 7.12.2, 16-bit A layout).
__device__ __forceinline__ v16bf loadA(const __bf16* buf, int row, int kt, int kb0) {
  const __bf16* base = buf + row * NH + kt * 32 + kb0;
  u32x4 a = *(const u32x4*)(base);
  u32x4 b = *(const u32x4*)(base + 16);
  return pack_bf(a, b);
}

// Same A fragment with inverted dropout fused in: ys0 = (u>=P) ? h/(1-P) : 0.
// urow points at u[t][b][0] (contiguous in K).
__device__ __forceinline__ v16bf loadA_drop(const __bf16* buf, int row, int kt, int kb0,
                                            const float* __restrict__ urow) {
  v16bf h = loadA(buf, row, kt, kb0);
  v16bf r;
#pragma unroll
  for (int e = 0; e < 8; ++e) {
    int k0 = kt * 32 + kb0 + e;
    int k1 = k0 + 16;
    float h0 = (float)h[e] * kPinv;
    float h1 = (float)h[e + 8] * kPinv;
    r[e]     = (__bf16)((urow[k0] >= kP) ? h0 : 0.0f);
    r[e + 8] = (__bf16)((urow[k1] >= kP) ? h1 : 0.0f);
  }
  return r;
}

// B-matrix fragment (32x16 bf16, K x N): lane&15 = column N (gate col),
// lane half selects K groups {0..7,16..23} vs {8..15,24..31} (matches A).
// W row-major [gate][k]: B[k][n] = W[gcol][k]  (computes h @ W.T).
__device__ __forceinline__ v16bf loadB(const __bf16* W, int gcol, int kt, int hi) {
  const __bf16* base = W + gcol * NH + kt * 32 + hi * 8;
  u32x4 a = *(const u32x4*)(base);
  u32x4 b = *(const u32x4*)(base + 16);
  return pack_bf(a, b);
}

__device__ __forceinline__ v8f wmma_bf16(v16bf a, v16bf b, v8f c) {
  return __builtin_amdgcn_wmma_f32_16x16x32_bf16(false, a, false, b, (short)0, c,
                                                 false, false);
}

// ---------------------------------------------------------------------------
// K0: one-time bf16 conversion of layer-1 weights (kept L2-resident; streamed
// as B-fragments by every workgroup every step).
// ---------------------------------------------------------------------------
__global__ void convert_weights(const float* __restrict__ Wih1,
                                const float* __restrict__ Whh1,
                                __bf16* __restrict__ wih1b,
                                __bf16* __restrict__ whh1b) {
  int i = blockIdx.x * blockDim.x + threadIdx.x;
  if (i < NG * NH) {
    wih1b[i] = (__bf16)Wih1[i];
    whh1b[i] = (__bf16)Whh1[i];
  }
}

// ---------------------------------------------------------------------------
// Fused kernel: layer0 scan -> (fused enc dropout) -> layer1 scan -> hn dropout
// -> 12-step decoder with dec dropout + linear head, out fed back as input.
// ---------------------------------------------------------------------------
__global__ __launch_bounds__(NTHR)
void mcd_lstm_fused(const float* __restrict__ obs,
                    const float* __restrict__ Wih0, const float* __restrict__ Whh0,
                    const float* __restrict__ bih0, const float* __restrict__ bhh0,
                    const __bf16* __restrict__ Wih1b, const __bf16* __restrict__ Whh1b,
                    const float* __restrict__ bih1, const float* __restrict__ bhh1,
                    const float* __restrict__ Wihc, const float* __restrict__ Whhc,
                    const float* __restrict__ bihc, const float* __restrict__ bhhc,
                    const float* __restrict__ Wout, const float* __restrict__ bout,
                    const float* __restrict__ enc_u, const float* __restrict__ hn_u,
                    const float* __restrict__ dec_u,
                    float* __restrict__ out) {
  // LDS: 128K weights + 2x16K state = 160 KB (CU-mode safe on a 320KB WGP).
  __shared__ __align__(16) __bf16 Wlds[NG * NH];   // Whh0 (encoder) / Whhc (decoder)
  __shared__ __align__(16) __bf16 h0buf[BT * NH];  // layer0 h (undropped); decoder scratch
  __shared__ __align__(16) __bf16 h1buf[BT * NH];  // layer1 / decoder h

  const int tid   = threadIdx.x;
  const int lane  = tid & 31;
  const int wave  = tid >> 5;    // 0..15
  const int rt    = wave >> 2;   // row tile 0..3 (16 rows each)
  const int cq    = wave & 3;    // gate-column pair: jb = cq*2 + jl
  const int n     = lane & 15;   // tile column (C/D layout, B layout)
  const int hi    = lane >> 4;   // lane half
  const int kb0   = hi * 8;      // A-fragment K base
  const int mB    = hi * 8;      // C/D row offset (rows = mB + v)
  const int bBase = blockIdx.x * BT;
  const int rowA  = rt * 16 + n; // row for A-fragment loads

  // Stage Whh0 -> LDS bf16; zero initial h state.
  for (int i = tid; i < NG * NH; i += NTHR) Wlds[i] = (__bf16)Whh0[i];
  for (int i = tid; i < BT * NH; i += NTHR) {
    h0buf[i] = (__bf16)0.0f;
    h1buf[i] = (__bf16)0.0f;
  }

  // Loop-invariant per-lane scalars (hoisted out of the time loop).
  float bb0[NJB][4], wx0[NJB][4], wx1[NJB][4], bb1[NJB][4];
#pragma unroll
  for (int jl = 0; jl < NJB; ++jl) {
#pragma unroll
    for (int q = 0; q < 4; ++q) {
      int g = q * NH + (cq * NJB + jl) * 16 + n;
      bb0[jl][q] = bih0[g] + bhh0[g];
      wx0[jl][q] = Wih0[g * ND + 0];
      wx1[jl][q] = Wih0[g * ND + 1];
      bb1[jl][q] = bih1[g] + bhh1[g];
    }
  }

  v8f c0[NJB], c1[NJB];  // cell states, C/D layout tiles
#pragma unroll
  for (int j = 0; j < NJB; ++j) {
#pragma unroll
    for (int v = 0; v < 8; ++v) { c0[j][v] = 0.0f; c1[j][v] = 0.0f; }
  }
  __syncthreads();

  // ================= encoder: layer0 + layer1, fused per time step ==========
  for (int t = 0; t < NT; ++t) {
    // ---- layer0 A fragments (recurrent h0) + x for C-init ----
    v16bf a0[4];
#pragma unroll
    for (int kt = 0; kt < 4; ++kt) a0[kt] = loadA(h0buf, rowA, kt, kb0);
    float x0v[8], x1v[8];
#pragma unroll
    for (int v = 0; v < 8; ++v) {
      int b = bBase + rt * 16 + mB + v;
      const float* xp = obs + ((size_t)b * NT + t) * ND;
      x0v[v] = xp[0];
      x1v[v] = xp[1];
    }
    __syncthreads();

    // ---- layer0: gates = x@Wih0.T + h0@Whh0.T + b  (B-frags from LDS) ----
#pragma unroll 1
    for (int jl = 0; jl < NJB; ++jl) {
      int jb = cq * NJB + jl;
      int gb = jb * 16 + n;
      v8f acc[4];
#pragma unroll
      for (int q = 0; q < 4; ++q) {  // i,f,g,o gate groups
        v8f c;
#pragma unroll
        for (int v = 0; v < 8; ++v)
          c[v] = bb0[jl][q] + x0v[v] * wx0[jl][q] + x1v[v] * wx1[jl][q];
        v16bf bf[4];
#pragma unroll
        for (int kt = 0; kt < 4; ++kt) bf[kt] = loadB(Wlds, q * NH + gb, kt, hi);
#pragma unroll
        for (int kt = 0; kt < 4; ++kt) c = wmma_bf16(a0[kt], bf[kt], c);
        acc[q] = c;
      }
      int col = jb * 16 + n;
      v8f cn, hv8;
#pragma unroll
      for (int v = 0; v < 8; ++v) {
        float iv = sig_(acc[0][v]);
        float fv = sig_(acc[1][v]);
        float gv = tanhf(acc[2][v]);
        float ov = sig_(acc[3][v]);
        float cv = fv * c0[jl][v] + iv * gv;
        cn[v]  = cv;
        hv8[v] = ov * tanhf(cv);
      }
      c0[jl] = cn;
#pragma unroll
      for (int v = 0; v < 8; ++v)
        h0buf[(rt * 16 + mB + v) * NH + col] = (__bf16)hv8[v];
    }
    __syncthreads();

    // ---- layer1 A fragments: dropped ys0 (fused dropout) + recurrent h1 ----
    const float* urow = enc_u + ((size_t)t * NB + (bBase + rowA)) * NH;
    v16bf ay[4], a1[4];
#pragma unroll
    for (int kt = 0; kt < 4; ++kt) {
      ay[kt] = loadA_drop(h0buf, rowA, kt, kb0, urow);
      a1[kt] = loadA(h1buf, rowA, kt, kb0);
    }
    __syncthreads();

    // ---- layer1: gates = ys0@Wih1.T + h1@Whh1.T + b  (K=256, B from L2) ----
#pragma unroll 1
    for (int jl = 0; jl < NJB; ++jl) {
      int jb = cq * NJB + jl;
      int gb = jb * 16 + n;
      v8f acc[4];
#pragma unroll
      for (int q = 0; q < 4; ++q) {
        float bb = bb1[jl][q];
        v8f c = {bb, bb, bb, bb, bb, bb, bb, bb};
        v16bf bf[4];
#pragma unroll
        for (int kt = 0; kt < 4; ++kt) bf[kt] = loadB(Wih1b, q * NH + gb, kt, hi);
#pragma unroll
        for (int kt = 0; kt < 4; ++kt) c = wmma_bf16(ay[kt], bf[kt], c);
#pragma unroll
        for (int kt = 0; kt < 4; ++kt) bf[kt] = loadB(Whh1b, q * NH + gb, kt, hi);
#pragma unroll
        for (int kt = 0; kt < 4; ++kt) c = wmma_bf16(a1[kt], bf[kt], c);
        acc[q] = c;
      }
      int col = jb * 16 + n;
      v8f cn, hv8;
#pragma unroll
      for (int v = 0; v < 8; ++v) {
        float iv = sig_(acc[0][v]);
        float fv = sig_(acc[1][v]);
        float gv = tanhf(acc[2][v]);
        float ov = sig_(acc[3][v]);
        float cv = fv * c1[jl][v] + iv * gv;
        cn[v]  = cv;
        hv8[v] = ov * tanhf(cv);
      }
      c1[jl] = cn;
#pragma unroll
      for (int v = 0; v < 8; ++v)
        h1buf[(rt * 16 + mB + v) * NH + col] = (__bf16)hv8[v];
    }
    __syncthreads();
  }

  // ================= decoder prologue ======================================
  // Restage Whhc into the weight LDS region; apply hn dropout to h1 in place.
  for (int i = tid; i < NG * NH; i += NTHR) Wlds[i] = (__bf16)Whhc[i];
  for (int i = tid; i < BT * NH; i += NTHR) {
    int r = i >> 7, k = i & (NH - 1);
    float u = hn_u[(size_t)(bBase + r) * NH + k];
    float h = (float)h1buf[i];
    h1buf[i] = (__bf16)((u >= kP) ? h * kPinv : 0.0f);
  }
  // Decoder per-lane invariants (encoder regs are dead now).
  float bbc[NJB][4], wc0[NJB][4], wc1[NJB][4], wo0[NJB], wo1[NJB];
#pragma unroll
  for (int jl = 0; jl < NJB; ++jl) {
#pragma unroll
    for (int q = 0; q < 4; ++q) {
      int g = q * NH + (cq * NJB + jl) * 16 + n;
      bbc[jl][q] = bihc[g] + bhhc[g];
      wc0[jl][q] = Wihc[g * ND + 0];
      wc1[jl][q] = Wihc[g * ND + 1];
    }
    int col = (cq * NJB + jl) * 16 + n;
    wo0[jl] = Wout[0 * NH + col];
    wo1[jl] = Wout[1 * NH + col];
  }
  // h0buf region is dead: reuse for decoder input & output accumulators.
  float* inpBuf = (float*)h0buf;     // [BT][NO]
  float* outAcc = inpBuf + BT * NO;  // [BT][NO]
  if (tid < BT * NO) {
    int r = tid >> 1, d = tid & 1;
    inpBuf[tid] = obs[((size_t)(bBase + r) * NT + (NT - 1)) * ND + d];
  }
  __syncthreads();

  // ================= decoder: 12 autoregressive steps ======================
  for (int p = 0; p < NPL; ++p) {
    if (tid < BT * NO) outAcc[tid] = 0.0f;
    v16bf ad[4];
#pragma unroll
    for (int kt = 0; kt < 4; ++kt) ad[kt] = loadA(h1buf, rowA, kt, kb0);
    float i0v[8], i1v[8];
#pragma unroll
    for (int v = 0; v < 8; ++v) {
      int r = rt * 16 + mB + v;
      i0v[v] = inpBuf[r * 2 + 0];
      i1v[v] = inpBuf[r * 2 + 1];
    }
    __syncthreads();

    float po0[8], po1[8];
#pragma unroll
    for (int v = 0; v < 8; ++v) { po0[v] = 0.0f; po1[v] = 0.0f; }

#pragma unroll 1
    for (int jl = 0; jl < NJB; ++jl) {
      int jb = cq * NJB + jl;
      int gb = jb * 16 + n;
      v8f acc[4];
#pragma unroll
      for (int q = 0; q < 4; ++q) {
        v8f c;
#pragma unroll
        for (int v = 0; v < 8; ++v)
          c[v] = bbc[jl][q] + i0v[v] * wc0[jl][q] + i1v[v] * wc1[jl][q];
        v16bf bf[4];
#pragma unroll
        for (int kt = 0; kt < 4; ++kt) bf[kt] = loadB(Wlds, q * NH + gb, kt, hi);
#pragma unroll
        for (int kt = 0; kt < 4; ++kt) c = wmma_bf16(ad[kt], bf[kt], c);
        acc[q] = c;
      }
      int col = jb * 16 + n;
      v8f cn;
#pragma unroll
      for (int v = 0; v < 8; ++v) {
        float iv = sig_(acc[0][v]);
        float fv = sig_(acc[1][v]);
        float gv = tanhf(acc[2][v]);
        float ov = sig_(acc[3][v]);
        float cv = fv * c1[jl][v] + iv * gv;
        float hv = ov * tanhf(cv);
        int b = bBase + rt * 16 + mB + v;
        float u = dec_u[((size_t)p * NB + b) * NH + col];
        float hd = (u >= kP) ? hv * kPinv : 0.0f;  // carried h is the dropped h
        cn[v] = cv;
        h1buf[(rt * 16 + mB + v) * NH + col] = (__bf16)hd;
        po0[v] += hd * wo0[jl];  // linear head partials (reduce over col)
        po1[v] += hd * wo1[jl];
      }
      c1[jl] = cn;
    }
#pragma unroll
    for (int v = 0; v < 8; ++v) {
      int r = rt * 16 + mB + v;
      atomicAdd(&outAcc[r * 2 + 0], po0[v]);
      atomicAdd(&outAcc[r * 2 + 1], po1[v]);
    }
    __syncthreads();
    if (tid < BT * NO) {
      int r = tid >> 1, o = tid & 1;
      float val = outAcc[tid] + bout[o];
      out[((size_t)(bBase + r) * NPL + p) * NO + o] = val;  // (B, PL, O)
      inpBuf[tid] = val;  // out feeds back as next decoder input
    }
    __syncthreads();
  }
}

// ---------------------------------------------------------------------------
extern "C" void kernel_launch(void* const* d_in, const int* in_sizes, int n_in,
                              void* d_out, int out_size, void* d_ws, size_t ws_size,
                              hipStream_t stream) {
  (void)in_sizes; (void)n_in; (void)out_size; (void)ws_size;
  const float* obs   = (const float*)d_in[0];
  const float* Wih0  = (const float*)d_in[1];
  const float* Whh0  = (const float*)d_in[2];
  const float* bih0  = (const float*)d_in[3];
  const float* bhh0  = (const float*)d_in[4];
  const float* Wih1  = (const float*)d_in[5];
  const float* Whh1  = (const float*)d_in[6];
  const float* bih1  = (const float*)d_in[7];
  const float* bhh1  = (const float*)d_in[8];
  const float* Wihc  = (const float*)d_in[9];
  const float* Whhc  = (const float*)d_in[10];
  const float* bihc  = (const float*)d_in[11];
  const float* bhhc  = (const float*)d_in[12];
  const float* Wout  = (const float*)d_in[13];
  const float* bout  = (const float*)d_in[14];
  const float* enc_u = (const float*)d_in[15];
  const float* hn_u  = (const float*)d_in[16];
  const float* dec_u = (const float*)d_in[17];

  __bf16* wih1b = (__bf16*)d_ws;    // 128 KB
  __bf16* whh1b = wih1b + NG * NH;  // 128 KB

  convert_weights<<<(NG * NH + 255) / 256, 256, 0, stream>>>(Wih1, Whh1, wih1b, whh1b);
  mcd_lstm_fused<<<NB / BT, NTHR, 0, stream>>>(
      obs, Wih0, Whh0, bih0, bhh0, wih1b, whh1b, bih1, bhh1,
      Wihc, Whhc, bihc, bhhc, Wout, bout, enc_u, hn_u, dec_u, (float*)d_out);
}
